// BidirectionalTransformer_62405874811787
// MI455X (gfx1250) — matrix operand order repaired and checked by reading, not compiled, not run on hardware
//
#include <hip/hip_runtime.h>
#include <hip/hip_bf16.h>
#include <math.h>

// ---- problem constants ----
#define IMG    32
#define WSZ    8
#define SHIFTV 4
#define CDIM   512
#define NHEADS 16
#define DEPTH  24
#define EDIM   256
#define NE     8192
#define BATCH  8
#define LTOK   1024
#define NWTOK  64           // tokens per window
#define HD     32           // head dim
#define MROWS  (BATCH * LTOK)  // 8192

typedef __attribute__((ext_vector_type(16))) __bf16          v16bf;
typedef __attribute__((ext_vector_type(8)))  float           v8f;
typedef __attribute__((ext_vector_type(8)))  unsigned short  ushort8;

// fp32 -> bf16 round-to-nearest-even
__device__ inline unsigned int f2bf_u(float f) {
  unsigned int u = __builtin_bit_cast(unsigned int, f);
  u += 0x7fffu + ((u >> 16) & 1u);
  return u >> 16;
}
__device__ inline unsigned short f2bf(float f) {
  return (unsigned short)f2bf_u(f);
}
// pack two fp32 -> one u32 holding 2 bf16 (lo, hi)
__device__ inline unsigned int f2bf2(float lo, float hi) {
  return f2bf_u(lo) | (f2bf_u(hi) << 16);
}

__device__ inline float gelu_f(float v) {
  return 0.5f * v * (1.f + erff(v * 0.7071067811865475f));
}

// A fragment (16x32 bf16, M x K), per CDNA5 ISA 7.12.2:
// lane<16:  halves[0..7]=K0..7,  halves[8..15]=K16..23 of row M=lane
// lane>=16: halves[0..7]=K8..15, halves[8..15]=K24..31 of row M=lane-16
__device__ inline v16bf load_frag_a(const unsigned short* base, int ld, int row0,
                                    int koff, int lane) {
  int r  = row0 + (lane & 15);
  int kb = koff + ((lane < 16) ? 0 : 8);
  union { v16bf v; ushort8 h[2]; } u;
  u.h[0] = *(const ushort8*)(base + r * ld + kb);
  u.h[1] = *(const ushort8*)(base + r * ld + kb + 16);
  return u.v;
}

// B fragment (32x16 bf16, K x N), B stored as [N][K] contiguous-in-K:
// lane<16: column N=lane, K=0..7,16..23 ; lane>=16: column N=lane-16, K=8..15,24..31
__device__ inline v16bf load_frag_b(const unsigned short* base, int ld, int col0,
                                    int koff, int lane) {
  int n  = col0 + (lane & 15);
  int kb = koff + ((lane < 16) ? 0 : 8);
  const ushort8* p = (const ushort8*)(base + n * ld + kb);
  union { v16bf v; ushort8 h[2]; } u;
  u.h[0] = p[0];
  u.h[1] = p[2];
  return u.v;
}

__device__ inline v8f wmma_bf16(v16bf a, v16bf b, v8f c) {
  return __builtin_amdgcn_wmma_f32_16x16x32_bf16(false, a, false, b, (short)0, c,
                                                 false, false);
}

// ------------------------------------------------------------------
// Generic GEMM: C[M,N] = A[M,K] @ W[N,K]^T + bias, fp32 io, bf16 WMMA math.
// Software pipelined: next K-tile prefetched (float2 / b64 loads) into
// registers while the current tile's WMMAs execute; staging packs two bf16
// per ds_store_b32.
// epi: 0 = bias only, 1 = bias + exact GELU, 2 = bias + residual add (res)
// M % 128 == 0, N % 128 == 0, K % 32 == 0 (guaranteed by our shapes).
// ------------------------------------------------------------------
#define BM  128
#define BN  128
#define BKK 32
#define LDT 40   // padded LDS row stride (elements); 80B keeps 16B alignment

__global__ __launch_bounds__(256) void wmma_gemm(
    const float* __restrict__ A, const float* __restrict__ W,
    const float* __restrict__ bias, const float* __restrict__ res,
    float* __restrict__ C, int M, int N, int K, int epi) {
  __shared__ unsigned short As[BM * LDT];
  __shared__ unsigned short Bs[BN * LDT];

  const int tid  = threadIdx.x;
  const int lane = tid & 31;
  const int wid  = tid >> 5;   // 0..7
  const int wm   = wid >> 2;   // 0..1  -> 64 rows each
  const int wn   = wid & 3;    // 0..3  -> 32 cols each
  const int mBase = blockIdx.y * BM;
  const int nBase = blockIdx.x * BN;

  v8f zero8;
  for (int r = 0; r < 8; ++r) zero8[r] = 0.f;
  v8f acc[4][2];
  for (int i = 0; i < 4; ++i)
    for (int j = 0; j < 2; ++j) acc[i][j] = zero8;

  // 8 column-pairs per matrix per thread (128x32 tile = 2048 pairs / 256 thr)
  float2 aReg[8], bReg[8];

#pragma unroll
  for (int i = 0; i < 8; ++i) {
    int p = tid + 256 * i;          // pair index
    int r = p >> 4, c = (p & 15) * 2;
    aReg[i] = *(const float2*)(A + (size_t)(mBase + r) * K + c);
    bReg[i] = *(const float2*)(W + (size_t)(nBase + r) * K + c);
  }

  for (int k0 = 0; k0 < K; k0 += BKK) {
    // stage current tile: pack 2x bf16 per 32-bit LDS store
#pragma unroll
    for (int i = 0; i < 8; ++i) {
      int p = tid + 256 * i;
      int r = p >> 4, c = (p & 15) * 2;
      *(unsigned int*)(As + r * LDT + c) = f2bf2(aReg[i].x, aReg[i].y);
      *(unsigned int*)(Bs + r * LDT + c) = f2bf2(bReg[i].x, bReg[i].y);
    }
    __syncthreads();

    // issue global loads for the NEXT tile; they retire while WMMAs run
    if (k0 + BKK < K) {
      int kn = k0 + BKK;
#pragma unroll
      for (int i = 0; i < 8; ++i) {
        int p = tid + 256 * i;
        int r = p >> 4, c = (p & 15) * 2;
        aReg[i] = *(const float2*)(A + (size_t)(mBase + r) * K + kn + c);
        bReg[i] = *(const float2*)(W + (size_t)(nBase + r) * K + kn + c);
      }
    }

    v16bf b0 = load_frag_b(Bs, LDT, wn * 32 + 0,  0, lane);
    v16bf b1 = load_frag_b(Bs, LDT, wn * 32 + 16, 0, lane);
#pragma unroll
    for (int i = 0; i < 4; ++i) {
      v16bf a = load_frag_a(As, LDT, wm * 64 + 16 * i, 0, lane);
      acc[i][0] = wmma_bf16(a, b0, acc[i][0]);
      acc[i][1] = wmma_bf16(a, b1, acc[i][1]);
    }
    __syncthreads();
  }

  // epilogue: C layout per ISA: VGPR r -> M=r (+8 for lanes 16..31), N=lane&15
  const int mo = (lane < 16) ? 0 : 8;
#pragma unroll
  for (int i = 0; i < 4; ++i) {
#pragma unroll
    for (int j = 0; j < 2; ++j) {
      int mrow = mBase + wm * 64 + 16 * i + mo;
      int ncol = nBase + wn * 32 + 16 * j + (lane & 15);
      float bv = bias[ncol];
#pragma unroll
      for (int r = 0; r < 8; ++r) {
        float v = acc[i][j][r] + bv;
        if (epi == 1) v = gelu_f(v);
        size_t idx = (size_t)(mrow + r) * N + ncol;
        if (epi == 2) v += res[idx];
        C[idx] = v;
      }
    }
  }
}

// ------------------------------------------------------------------
// LayerNorm over last dim (512), one block per row; optional fused GELU
// ------------------------------------------------------------------
__global__ __launch_bounds__(256) void ln_kernel(const float* __restrict__ x,
                                                 const float* __restrict__ w,
                                                 const float* __restrict__ b,
                                                 float* __restrict__ out,
                                                 int applyGelu) {
  __shared__ float red[256];
  const int row = blockIdx.x;
  const int tid = threadIdx.x;
  const float* xr = x + (size_t)row * CDIM;
  float a0 = xr[tid], a1 = xr[tid + 256];

  red[tid] = a0 + a1;
  __syncthreads();
  for (int o = 128; o > 0; o >>= 1) {
    if (tid < o) red[tid] += red[tid + o];
    __syncthreads();
  }
  float mean = red[0] * (1.f / CDIM);
  __syncthreads();

  float d0 = a0 - mean, d1 = a1 - mean;
  red[tid] = d0 * d0 + d1 * d1;
  __syncthreads();
  for (int o = 128; o > 0; o >>= 1) {
    if (tid < o) red[tid] += red[tid + o];
    __syncthreads();
  }
  float rstd = rsqrtf(red[0] * (1.f / CDIM) + 1e-5f);

  float o0 = d0 * rstd * w[tid] + b[tid];
  float o1 = d1 * rstd * w[tid + 256] + b[tid + 256];
  if (applyGelu) { o0 = gelu_f(o0); o1 = gelu_f(o1); }
  float* orow = out + (size_t)row * CDIM;
  orow[tid] = o0;
  orow[tid + 256] = o1;
}

// x += pos_embed (broadcast over batch)
__global__ void add_pos_kernel(float* __restrict__ x, const float* __restrict__ pos) {
  int e = blockIdx.x * 256 + threadIdx.x;
  x[e] += pos[e & (LTOK * CDIM - 1)];
}

// roll(-shift) + window partition: hw[(b*16+win)*64+t][c] = h[b, y, x, c]
__global__ void win_gather_kernel(const float* __restrict__ h, float* __restrict__ hw,
                                  int shift) {
  int e = blockIdx.x * 256 + threadIdx.x;
  int c   = e & (CDIM - 1);
  int row = e >> 9;               // output row
  int t   = row & 63;
  int win = (row >> 6) & 15;
  int b   = row >> 10;
  int wy = win >> 2, wx = win & 3;
  int ty = t >> 3,   tx = t & 7;
  int y  = (wy * 8 + ty + shift) & 31;
  int xx = (wx * 8 + tx + shift) & 31;
  int srow = b * LTOK + y * IMG + xx;
  hw[e] = h[(size_t)srow * CDIM + c];
}

// window reverse + roll(+shift) + residual add into x
__global__ void win_scatter_add_kernel(const float* __restrict__ p, float* __restrict__ x,
                                       int shift) {
  int e = blockIdx.x * 256 + threadIdx.x;
  int c    = e & (CDIM - 1);
  int orow = e >> 9;              // b*1024 + y*32 + x
  int b   = orow >> 10;
  int pix = orow & 1023;
  int y = pix >> 5, xx = pix & 31;
  int yy = (y - shift) & 31;
  int xs = (xx - shift) & 31;
  int win = (yy >> 3) * 4 + (xs >> 3);
  int t   = (yy & 7) * 8 + (xs & 7);
  int srow = (b * 16 + win) * NWTOK + t;
  x[e] += p[(size_t)srow * CDIM + c];
}

// ------------------------------------------------------------------
// Windowed attention: one wave per (head, window).
// qkv: [8192, 1536] window-layout rows. out: [8192, 512] window layout.
// ------------------------------------------------------------------
__device__ inline int regidx(int v) {
  return (v < IMG - WSZ) ? 0 : ((v < IMG - SHIFTV) ? 1 : 2);
}

__global__ __launch_bounds__(32) void attn_kernel(const float* __restrict__ qkv,
                                                  const float* __restrict__ rel_bias,
                                                  float* __restrict__ out, int shift) {
  __shared__ unsigned short Qs[64 * 40];   // [token][dim] bf16 (scaled Q)
  __shared__ unsigned short Ks[64 * 40];   // [token][dim]
  __shared__ unsigned short Vt[32 * 72];   // [dim][token] (transposed V)
  __shared__ float          Sc[64 * 66];   // scores fp32
  __shared__ unsigned short Ps[64 * 72];   // softmax probs bf16

  const int lane   = threadIdx.x;
  const int head   = blockIdx.x;           // 0..15
  const int win    = blockIdx.y;           // 0..127
  const int wloc   = win & 15;
  const float scale = 0.17677669529663687f;  // 1/sqrt(32)
  const size_t rowbase = (size_t)win * NWTOK;
  const float* rb = rel_bias + head;       // rel_bias[idx][head]

  // Q/K staged as packed bf16 pairs; V staged transposed (scalar stores)
  for (int idx = lane; idx < NWTOK * (HD / 2); idx += 32) {
    int t = idx >> 4, d = (idx & 15) * 2;
    const float* r = qkv + (rowbase + t) * (3 * CDIM);
    float2 q = *(const float2*)(r + head * HD + d);
    float2 k = *(const float2*)(r + CDIM + head * HD + d);
    float2 v = *(const float2*)(r + 2 * CDIM + head * HD + d);
    *(unsigned int*)(Qs + t * 40 + d) = f2bf2(q.x * scale, q.y * scale);
    *(unsigned int*)(Ks + t * 40 + d) = f2bf2(k.x, k.y);
    Vt[d * 72 + t]       = f2bf(v.x);
    Vt[(d + 1) * 72 + t] = f2bf(v.y);
  }
  __syncthreads();

  v8f zero8;
  for (int r = 0; r < 8; ++r) zero8[r] = 0.f;

  // S = (Q*scale) @ K^T  : 4x4 tiles of 16x16, single K-step of 32
  const int mo = (lane < 16) ? 0 : 8;
#pragma unroll
  for (int i = 0; i < 4; ++i) {
    v16bf a = load_frag_a(Qs, 40, 16 * i, 0, lane);
#pragma unroll
    for (int j = 0; j < 4; ++j) {
      v16bf b = load_frag_b(Ks, 40, 16 * j, 0, lane);
      v8f s = wmma_bf16(a, b, zero8);
      int n = 16 * j + (lane & 15);
#pragma unroll
      for (int r = 0; r < 8; ++r) Sc[(16 * i + mo + r) * 66 + n] = s[r];
    }
  }
  __syncthreads();

  // bias + mask + softmax; each lane owns 2 rows
  const int wy = wloc >> 2, wx = wloc & 3;
  for (int rr = 0; rr < 2; ++rr) {
    int m = lane * 2 + rr;
    int my = m >> 3, mx_ = m & 7;
    int rm = regidx(wy * 8 + my) * 3 + regidx(wx * 8 + mx_);
    float* row = &Sc[m * 66];
    float mv = -1e30f;
    for (int n = 0; n < 64; ++n) {
      int ny = n >> 3, nx = n & 7;
      int bidx = (my - ny + 7) * 15 + (mx_ - nx + 7);
      float v = row[n] + rb[bidx * NHEADS];
      if (shift) {
        int rn = regidx(wy * 8 + ny) * 3 + regidx(wx * 8 + nx);
        if (rn != rm) v -= 100.f;
      }
      row[n] = v;
      mv = fmaxf(mv, v);
    }
    float sum = 0.f;
    for (int n = 0; n < 64; ++n) {
      float ev = expf(row[n] - mv);
      sum += ev;
      row[n] = ev;
    }
    float inv = 1.f / sum;
    for (int n = 0; n < 32; ++n)
      *(unsigned int*)(Ps + m * 72 + 2 * n) =
          f2bf2(row[2 * n] * inv, row[2 * n + 1] * inv);
  }
  __syncthreads();

  // O = P @ V : P is [64 x 64], V^T staged as [dim][token]
#pragma unroll
  for (int i = 0; i < 4; ++i) {
#pragma unroll
    for (int j = 0; j < 2; ++j) {
      v8f acc = zero8;
#pragma unroll
      for (int kk = 0; kk < 2; ++kk) {
        v16bf a = load_frag_a(Ps, 72, 16 * i, 32 * kk, lane);
        v16bf b = load_frag_b(Vt, 72, 16 * j, 32 * kk, lane);
        acc = wmma_bf16(a, b, acc);
      }
      int nn = 16 * j + (lane & 15);
#pragma unroll
      for (int r = 0; r < 8; ++r)
        out[(rowbase + 16 * i + mo + r) * CDIM + head * HD + nn] = acc[r];
    }
  }
}

// ------------------------------------------------------------------
// host orchestration
// ------------------------------------------------------------------
extern "C" void kernel_launch(void* const* d_in, const int* in_sizes, int n_in,
                              void* d_out, int out_size, void* d_ws, size_t ws_size,
                              hipStream_t stream) {
  const float* x_in   = (const float*)d_in[0];
  const float* dec_w  = (const float*)d_in[1];
  const float* dec_b  = (const float*)d_in[2];
  const float* pos    = (const float*)d_in[3];
  const float* n1w    = (const float*)d_in[4];
  const float* n1b    = (const float*)d_in[5];
  const float* qkv_w  = (const float*)d_in[6];
  const float* qkv_b  = (const float*)d_in[7];
  const float* proj_w = (const float*)d_in[8];
  const float* proj_b = (const float*)d_in[9];
  const float* rel_b  = (const float*)d_in[10];
  const float* n2w    = (const float*)d_in[11];
  const float* n2b    = (const float*)d_in[12];
  const float* fc1_w  = (const float*)d_in[13];
  const float* fc1_b  = (const float*)d_in[14];
  const float* fc2_w  = (const float*)d_in[15];
  const float* fc2_b  = (const float*)d_in[16];
  const float* nfw    = (const float*)d_in[17];
  const float* nfb    = (const float*)d_in[18];
  const float* pred_w = (const float*)d_in[19];
  const float* pred_b = (const float*)d_in[20];
  float* outp = (float*)d_out;

  const size_t T = (size_t)MROWS * CDIM;   // 4M floats
  float* ws   = (float*)d_ws;
  float* xs   = ws;                        // [8192,512] running activations
  float* hbuf = xs + T;                    // LN outputs
  float* hw   = hbuf + T;                  // gathered windows
  float* qkvb = hw + T;                    // [8192,1536]
  float* att  = qkvb + (size_t)MROWS * 3 * CDIM;  // attention out (window layout)
  float* proj = att + T;                   // proj out (window layout)
  float* fc1b = proj + T;                  // [8192,2048]

  auto gemm = [&](const float* A, const float* W, const float* bias, const float* res,
                  float* Cp, int M, int N, int K, int epi) {
    dim3 g(N / BN, M / BM);
    wmma_gemm<<<g, 256, 0, stream>>>(A, W, bias, res, Cp, M, N, K, epi);
  };

  const int TOT = MROWS * CDIM;            // 4194304

  // decoder + positional embedding
  gemm(x_in, dec_w, dec_b, nullptr, xs, MROWS, CDIM, EDIM, 0);
  add_pos_kernel<<<TOT / 256, 256, 0, stream>>>(xs, pos);

  for (int i = 0; i < DEPTH; ++i) {
    int shift = (i & 1) ? SHIFTV : 0;
    // LN1 -> gather windows (with roll)
    ln_kernel<<<MROWS, 256, 0, stream>>>(xs, n1w + i * CDIM, n1b + i * CDIM, hbuf, 0);
    win_gather_kernel<<<TOT / 256, 256, 0, stream>>>(hbuf, hw, shift);
    // QKV
    gemm(hw, qkv_w + (size_t)i * 3 * CDIM * CDIM, qkv_b + (size_t)i * 3 * CDIM,
         nullptr, qkvb, MROWS, 3 * CDIM, CDIM, 0);
    // attention
    attn_kernel<<<dim3(NHEADS, BATCH * 16), 32, 0, stream>>>(
        qkvb, rel_b + (size_t)i * 225 * NHEADS, att, shift);
    // output projection
    gemm(att, proj_w + (size_t)i * CDIM * CDIM, proj_b + (size_t)i * CDIM,
         nullptr, proj, MROWS, CDIM, CDIM, 0);
    // window reverse + residual
    win_scatter_add_kernel<<<TOT / 256, 256, 0, stream>>>(proj, xs, shift);
    // MLP
    ln_kernel<<<MROWS, 256, 0, stream>>>(xs, n2w + i * CDIM, n2b + i * CDIM, hbuf, 0);
    gemm(hbuf, fc1_w + (size_t)i * 4 * CDIM * CDIM, fc1_b + (size_t)i * 4 * CDIM,
         nullptr, fc1b, MROWS, 4 * CDIM, CDIM, 1 /*GELU*/);
    gemm(fc1b, fc2_w + (size_t)i * CDIM * 4 * CDIM, fc2_b + (size_t)i * CDIM,
         xs, xs, MROWS, CDIM, 4 * CDIM, 2 /*residual*/);
  }

  // final LN + GELU, then prediction head -> d_out (fp32)
  ln_kernel<<<MROWS, 256, 0, stream>>>(xs, nfw, nfb, hbuf, 1);
  gemm(hbuf, pred_w, pred_b, nullptr, outp, MROWS, NE, CDIM, 0);
}
